// RGATransformerDecoderLayer_49091476194115
// MI455X (gfx1250) — matrix operand order repaired and checked by reading, not compiled, not run on hardware
//
#include <hip/hip_runtime.h>
#include <cstdint>

typedef _Float16 v8h  __attribute__((ext_vector_type(8)));
typedef _Float16 v16h __attribute__((ext_vector_type(16)));
typedef float    v8f  __attribute__((ext_vector_type(8)));
typedef int      v4i  __attribute__((ext_vector_type(4)));

#define DM   1024
#define FFND 4096
#define LQN  1024
#define LKN  1024
#define BN   4
#define HN   16
#define HD   64

// ---- gfx1250 async global->LDS path (guarded; falls back to ds stores) ----
#if __has_builtin(__builtin_amdgcn_global_load_async_to_lds_b128)
#define USE_ASYNC_LDS 1
#else
#define USE_ASYNC_LDS 0
#endif

#if USE_ASYNC_LDS
typedef v4i __attribute__((address_space(1)))* gptr128_t;
typedef v4i __attribute__((address_space(3)))* lptr128_t;
// Generic->AS1: full 64-bit address.  Generic->AS3: LDS aperture keeps the
// LDS byte offset in the low 32 bits (ISA 10.2 aperture rules).
__device__ inline void async_copy16(const void* g, void* l) {
  __builtin_amdgcn_global_load_async_to_lds_b128(
      (gptr128_t)(uintptr_t)g, (lptr128_t)(uint32_t)(uintptr_t)l, 0, 0);
}
__device__ inline void async_wait0() {
#if __has_builtin(__builtin_amdgcn_s_wait_asynccnt)
  __builtin_amdgcn_s_wait_asynccnt(0);
#else
  asm volatile("s_wait_asynccnt 0x0" ::: "memory");
#endif
}
#endif

__device__ inline v8f v8f_zero() {
  v8f z = {0.f,0.f,0.f,0.f,0.f,0.f,0.f,0.f};
  return z;
}

// A-fragment (16x32 f16): lane holds row M=lane&15; kg=lane>>4 selects
// K halves {kg*8..kg*8+7} and {16+kg*8..16+kg*8+7} (ISA 7.12.2).
__device__ inline v16h load_a_frag(const _Float16* rowp, int kg) {
  v8h lo = *(const v8h*)(rowp + kg * 8);
  v8h hi = *(const v8h*)(rowp + 16 + kg * 8);
  return __builtin_shufflevector(lo, hi, 0,1,2,3,4,5,6,7,8,9,10,11,12,13,14,15);
}

// B-fragment (32x16 f16), unused: 16 contiguous halves from p; lane L is column N=L&15.
__device__ inline v16h load_b_frag(const _Float16* p) {
  v8h lo = *(const v8h*)(p);
  v8h hi = *(const v8h*)(p + 8);
  return __builtin_shufflevector(lo, hi, 0,1,2,3,4,5,6,7,8,9,10,11,12,13,14,15);
}

// ---------------------------------------------------------------- convert
__global__ void cvt_f32_f16_kernel(const float* __restrict__ src,
                                   _Float16* __restrict__ dst, int n) {
  int i = blockIdx.x * blockDim.x + threadIdx.x;
  int stride = gridDim.x * blockDim.x;
  for (; i < n; i += stride) dst[i] = (_Float16)src[i];
}

// ------------------------------------------------------- rel-term stats
// rel_e[m,d] = d - m + 1024 (unclipped in range) =>
// rel[l,m] = relA[l] - m*relS[l]; relA = sum_d q*(d+1024), relS = sum_d q.
__global__ void relstats_kernel(const _Float16* __restrict__ q,
                                float* __restrict__ relA,
                                float* __restrict__ relS) {
  int idx = blockIdx.x * blockDim.x + threadIdx.x;  // b*16384 + h*1024 + l
  int l = idx & (LQN - 1);
  int h = (idx >> 10) & (HN - 1);
  int b = idx >> 14;
  const _Float16* p = q + ((size_t)b * LQN + l) * DM + h * HD;
  float sA = 0.f, sS = 0.f;
#pragma unroll
  for (int d = 0; d < HD; ++d) {
    float qv = (float)p[d];
    sS += qv;
    sA += qv * (float)(d + 1024);
  }
  relA[idx] = sA;
  relS[idx] = sS;
}

// ---------------------------------------------------------------- GEMM
// C[M,N] = A[M,K](f16) @ B[K,N](f16) + bias, opt residual(f32)/relu,
// out f32 and/or f16.  Block 256 (8 waves), tile 128x128, K-step 32.
template<bool RELU, bool RESID, bool OUT16, bool OUT32>
__global__ __launch_bounds__(256)
void gemm_wmma_kernel(const _Float16* __restrict__ A,
                      const _Float16* __restrict__ Bm,
                      const float* __restrict__ bias,
                      const float* __restrict__ resid,
                      float* __restrict__ C32, _Float16* __restrict__ C16,
                      int M, int N, int K) {
  __shared__ alignas(16) _Float16 As[128][40];
  __shared__ alignas(16) _Float16 Bs[32][136];

  const int tid = threadIdx.x;
  const int lane = tid & 31, w = tid >> 5;
  const int hi = lane >> 4, ln = lane & 15;
  const int wm = w >> 2, wn = w & 3;  // wave grid 2(M) x 4(N), tile 64x32
  const int mbase = blockIdx.y * 128;
  const int nbase = blockIdx.x * 128;

  v8f acc[4][2];
#pragma unroll
  for (int mt = 0; mt < 4; ++mt)
#pragma unroll
    for (int nt = 0; nt < 2; ++nt) acc[mt][nt] = v8f_zero();

  const int arow = tid >> 1, acol = (tid & 1) * 16;
  const int brow = tid >> 3, bcol = (tid & 7) * 16;
  const _Float16* Aptr = A + (size_t)(mbase + arow) * K + acol;
  const _Float16* Bptr = Bm + (size_t)brow * N + nbase + bcol;

  for (int k0 = 0; k0 < K; k0 += 32) {
    __syncthreads();
#if USE_ASYNC_LDS
    async_copy16(Aptr + k0, &As[arow][acol]);
    async_copy16(Aptr + k0 + 8, &As[arow][acol + 8]);
    async_copy16(Bptr + (size_t)k0 * N, &Bs[brow][bcol]);
    async_copy16(Bptr + (size_t)k0 * N + 8, &Bs[brow][bcol + 8]);
#else
    v8h a0 = *(const v8h*)(Aptr + k0);
    v8h a1 = *(const v8h*)(Aptr + k0 + 8);
    v8h b0 = *(const v8h*)(Bptr + (size_t)k0 * N);
    v8h b1 = *(const v8h*)(Bptr + (size_t)k0 * N + 8);
    *(v8h*)&As[arow][acol] = a0;
    *(v8h*)&As[arow][acol + 8] = a1;
    *(v8h*)&Bs[brow][bcol] = b0;
    *(v8h*)&Bs[brow][bcol + 8] = b1;
#endif
    if (k0 + 32 < K) {
      __builtin_prefetch(Aptr + k0 + 32, 0, 3);
      __builtin_prefetch(Bptr + (size_t)(k0 + 32) * N, 0, 3);
    }
#if USE_ASYNC_LDS
    async_wait0();
#endif
    __syncthreads();
    v16h af[4];
#pragma unroll
    for (int mt = 0; mt < 4; ++mt)
      af[mt] = load_a_frag(&As[wm * 64 + mt * 16 + ln][0], hi);
#pragma unroll
    for (int nt = 0; nt < 2; ++nt) {
      v16h bf;
#pragma unroll
      for (int s = 0; s < 16; ++s)
        bf[s] = Bs[(s < 8) ? (hi * 8 + s) : (16 + hi * 8 + s - 8)][wn * 32 + nt * 16 + ln];
#pragma unroll
      for (int mt = 0; mt < 4; ++mt)
        acc[mt][nt] = __builtin_amdgcn_wmma_f32_16x16x32_f16(
            false, af[mt], false, bf, (short)0, acc[mt][nt], false, false);
    }
  }

#pragma unroll
  for (int mt = 0; mt < 4; ++mt) {
#pragma unroll
    for (int nt = 0; nt < 2; ++nt) {
      const float* ap = (const float*)&acc[mt][nt];
#pragma unroll
      for (int r = 0; r < 8; ++r) {
        int row = mbase + wm * 64 + mt * 16 + r + 8 * hi;
        int col = nbase + wn * 32 + nt * 16 + ln;
        float v = ap[r] + bias[col];
        if (RESID) v += resid[(size_t)row * N + col];
        if (RELU) v = fmaxf(v, 0.f);
        if (OUT32) C32[(size_t)row * N + col] = v;
        if (OUT16) C16[(size_t)row * N + col] = (_Float16)v;
      }
    }
  }
}

// --------------------------------------------------- streaming attention
// Block = 128 q-rows x 1 head; 8 waves, each owns 16 q-rows.
// Online softmax over K/V tiles of 64 (K data == V data in this model).
// Skew term via relA/relS: f=(i+1)*1024+j, l=f/1025, c=f%1025,
// term = (c==0) ? 0 : relA[l] - (c-1)*relS[l].
template<bool CAUSAL>
__global__ __launch_bounds__(256)
void attn_wmma_kernel(const _Float16* __restrict__ qbuf,
                      const _Float16* __restrict__ kvbuf,
                      const float* __restrict__ relA,
                      const float* __restrict__ relS,
                      _Float16* __restrict__ obuf) {
  __shared__ alignas(16) _Float16 Qs[128][72];
  __shared__ alignas(16) _Float16 KTs[64][72];  // transposed: [d][j]
  __shared__ alignas(16) _Float16 Vs[64][72];   // [j][d]
  __shared__ alignas(16) _Float16 Ps[8][16][72];
  __shared__ float relAs[136], relSs[136];

  const int tid = threadIdx.x;
  const int lane = tid & 31, w = tid >> 5;
  const int hi = lane >> 4, ln = lane & 15;
  const int b = blockIdx.z, h = blockIdx.y;
  const int qbase = blockIdx.x * 128;

  const _Float16* qg = qbuf + ((size_t)b * LQN) * DM + h * HD;
  const _Float16* kg = kvbuf + ((size_t)b * LKN) * DM + h * HD;

  // Q tile 128x64 (1024 eight-half chunks)
  for (int c = tid; c < 1024; c += 256) {
    int row = c >> 3, col = (c & 7) * 8;
#if USE_ASYNC_LDS
    async_copy16(qg + (size_t)(qbase + row) * DM + col, &Qs[row][col]);
#else
    v8h val = *(const v8h*)(qg + (size_t)(qbase + row) * DM + col);
    *(v8h*)&Qs[row][col] = val;
#endif
  }
  {
    const float* rA = relA + (size_t)(b * HN + h) * LQN;
    const float* rS = relS + (size_t)(b * HN + h) * LQN;
    if (tid < 132) {
      int l = qbase + tid;
      float a = 0.f, s = 0.f;
      if (l < LQN) { a = rA[l]; s = rS[l]; }
      relAs[tid] = a;
      relSs[tid] = s;
    }
  }
#if USE_ASYNC_LDS
  async_wait0();
#endif
  __syncthreads();

  v16h aQ[2];
#pragma unroll
  for (int kk = 0; kk < 2; ++kk)
    aQ[kk] = load_a_frag(&Qs[w * 16 + ln][kk * 32], hi);

  v8f Od[4];
#pragma unroll
  for (int t = 0; t < 4; ++t) Od[t] = v8f_zero();
  float m_r[8], l_r[8];
#pragma unroll
  for (int r = 0; r < 8; ++r) { m_r[r] = -1e30f; l_r[r] = 0.f; }

  const int qrow0 = qbase + w * 16;
  const int jend = CAUSAL ? (qbase + 128) : LKN;

  for (int j0 = 0; j0 < jend; j0 += 64) {
    __syncthreads();
    // K/V tiles (same source data): straight copy to Vs, transposed to KTs
#pragma unroll
    for (int it = 0; it < 2; ++it) {
      int c = tid + it * 256;
      int row = c >> 3, col = (c & 7) * 8;
      v8h kvv = *(const v8h*)(kg + (size_t)(j0 + row) * DM + col);
      *(v8h*)&Vs[row][col] = kvv;
#pragma unroll
      for (int u = 0; u < 8; ++u) KTs[col + u][row] = kvv[u];
    }
    __syncthreads();

    // S = Q @ K^T  (16 x 64 per wave)
    v8f St[4];
#pragma unroll
    for (int t = 0; t < 4; ++t) {
      v8f acc = v8f_zero();
#pragma unroll
      for (int kk = 0; kk < 2; ++kk) {
        v16h bf = load_a_frag(&Vs[t * 16 + ln][kk * 32], hi);
        acc = __builtin_amdgcn_wmma_f32_16x16x32_f16(
            false, aQ[kk], false, bf, (short)0, acc, false, false);
      }
      St[t] = acc;
    }

    float* sp = (float*)St;
#pragma unroll
    for (int t = 0; t < 4; ++t) {
#pragma unroll
      for (int r = 0; r < 8; ++r) {
        int i = qrow0 + r + 8 * hi;
        int j = j0 + t * 16 + ln;
        unsigned f = (unsigned)(i + 1) * 1024u + (unsigned)j;
        unsigned l = f / 1025u;
        unsigned cc = f - l * 1025u;
        unsigned li = l - (unsigned)qbase;
        float term = (cc == 0u) ? 0.f
                                : (relAs[li] - (float)(cc - 1u) * relSs[li]);
        float s = (sp[t * 8 + r] + term) * 0.125f;  // 1/sqrt(64)
        if (CAUSAL && j > i) s = -1e9f;
        sp[t * 8 + r] = s;
      }
    }

    // online softmax (row reductions across each 16-lane group)
    float esc[8];
#pragma unroll
    for (int r = 0; r < 8; ++r) {
      float v0 = fmaxf(fmaxf(sp[r], sp[8 + r]), fmaxf(sp[16 + r], sp[24 + r]));
#pragma unroll
      for (int mk = 1; mk <= 8; mk <<= 1) v0 = fmaxf(v0, __shfl_xor(v0, mk, 32));
      float mn = fmaxf(m_r[r], v0);
      esc[r] = __expf(m_r[r] - mn);
      m_r[r] = mn;
    }
#pragma unroll
    for (int t = 0; t < 4; ++t)
#pragma unroll
      for (int r = 0; r < 8; ++r) sp[t * 8 + r] = __expf(sp[t * 8 + r] - m_r[r]);
#pragma unroll
    for (int r = 0; r < 8; ++r) {
      float s0 = sp[r] + sp[8 + r] + sp[16 + r] + sp[24 + r];
#pragma unroll
      for (int mk = 1; mk <= 8; mk <<= 1) s0 += __shfl_xor(s0, mk, 32);
      l_r[r] = l_r[r] * esc[r] + s0;
    }
    float* op = (float*)Od;
#pragma unroll
    for (int t = 0; t < 4; ++t)
#pragma unroll
      for (int r = 0; r < 8; ++r) op[t * 8 + r] *= esc[r];

    // P (C-layout) -> per-wave LDS -> A-layout (wave-local, LDS in-order)
#pragma unroll
    for (int t = 0; t < 4; ++t)
#pragma unroll
      for (int r = 0; r < 8; ++r)
        Ps[w][r + 8 * hi][t * 16 + ln] = (_Float16)sp[t * 8 + r];

    v16h aP[2];
#pragma unroll
    for (int kk = 0; kk < 2; ++kk)
      aP[kk] = load_a_frag(&Ps[w][ln][kk * 32], hi);
#pragma unroll
    for (int t = 0; t < 4; ++t) {
      v8f acc = Od[t];
#pragma unroll
      for (int kk = 0; kk < 2; ++kk) {
        v16h bf = load_a_frag(&KTs[t * 16 + ln][kk * 32], hi);
        acc = __builtin_amdgcn_wmma_f32_16x16x32_f16(
            false, aP[kk], false, bf, (short)0, acc, false, false);
      }
      Od[t] = acc;
    }
  }

  // normalize and store merged-head output (f16)
  _Float16* og = obuf + ((size_t)b * LQN) * DM + h * HD;
  float inv[8];
#pragma unroll
  for (int r = 0; r < 8; ++r) inv[r] = 1.f / l_r[r];
  const float* ofin = (const float*)Od;
#pragma unroll
  for (int t = 0; t < 4; ++t)
#pragma unroll
    for (int r = 0; r < 8; ++r) {
      int i = qrow0 + r + 8 * hi;
      int d = t * 16 + ln;
      og[(size_t)i * DM + d] = (_Float16)(ofin[t * 8 + r] * inv[r]);
    }
}

// -------------------------------------------------------------- layernorm
__global__ __launch_bounds__(256)
void layernorm_kernel(const float* __restrict__ x, const float* __restrict__ g,
                      const float* __restrict__ bta, float* __restrict__ y32,
                      _Float16* __restrict__ y16) {
  const int row = blockIdx.x;
  const int tid = threadIdx.x;
  __shared__ float red[256];
  const float* xr = x + (size_t)row * DM;
  float xv[4];
  float s = 0.f;
#pragma unroll
  for (int i = 0; i < 4; ++i) {
    xv[i] = xr[tid + i * 256];
    s += xv[i];
  }
  red[tid] = s;
  __syncthreads();
  for (int st = 128; st > 0; st >>= 1) {
    if (tid < st) red[tid] += red[tid + st];
    __syncthreads();
  }
  float mu = red[0] * (1.f / 1024.f);
  __syncthreads();
  float s2 = 0.f;
#pragma unroll
  for (int i = 0; i < 4; ++i) {
    float d = xv[i] - mu;
    s2 += d * d;
  }
  red[tid] = s2;
  __syncthreads();
  for (int st = 128; st > 0; st >>= 1) {
    if (tid < st) red[tid] += red[tid + st];
    __syncthreads();
  }
  float rstd = rsqrtf(red[0] * (1.f / 1024.f) + 1e-5f);
#pragma unroll
  for (int i = 0; i < 4; ++i) {
    int c = tid + i * 256;
    float y = (xv[i] - mu) * rstd * g[c] + bta[c];
    y32[(size_t)row * DM + c] = y;
    if (y16) y16[(size_t)row * DM + c] = (_Float16)y;
  }
}

// ------------------------------------------------------------------ host
extern "C" void kernel_launch(void* const* d_in, const int* in_sizes, int n_in,
                              void* d_out, int out_size, void* d_ws,
                              size_t ws_size, hipStream_t stream) {
  const float* tgt  = (const float*)d_in[0];
  const float* memx = (const float*)d_in[1];
  // d_in[2] = tgt_mask (causal tril; implemented analytically)
  const float* saWq = (const float*)d_in[3];
  const float* sabq = (const float*)d_in[4];
  const float* saWo = (const float*)d_in[5];
  const float* sabo = (const float*)d_in[6];
  const float* caWq = (const float*)d_in[7];
  const float* cabq = (const float*)d_in[8];
  const float* caWo = (const float*)d_in[9];
  const float* cabo = (const float*)d_in[10];
  const float* W1   = (const float*)d_in[11];
  const float* b1   = (const float*)d_in[12];
  const float* W2   = (const float*)d_in[13];
  const float* b2   = (const float*)d_in[14];
  const float* ln1g = (const float*)d_in[15];
  const float* ln1b = (const float*)d_in[16];
  const float* ln2g = (const float*)d_in[17];
  const float* ln2b = (const float*)d_in[18];
  const float* ln3g = (const float*)d_in[19];
  const float* ln3b = (const float*)d_in[20];
  float* out = (float*)d_out;
  (void)in_sizes; (void)n_in; (void)out_size; (void)ws_size;

  char* base = (char*)d_ws;
  size_t off = 0;
  auto alloc = [&](size_t bytes) -> char* {
    char* p = base + off;
    off = (off + bytes + 255) & ~(size_t)255;
    return p;
  };
  const size_t NTOK = (size_t)BN * LQN;  // 4096 rows

  _Float16* saWq16 = (_Float16*)alloc((size_t)DM * DM * 2);
  _Float16* saWo16 = (_Float16*)alloc((size_t)DM * DM * 2);
  _Float16* caWq16 = (_Float16*)alloc((size_t)DM * DM * 2);
  _Float16* caWo16 = (_Float16*)alloc((size_t)DM * DM * 2);
  _Float16* W1_16  = (_Float16*)alloc((size_t)DM * FFND * 2);
  _Float16* W2_16  = (_Float16*)alloc((size_t)FFND * DM * 2);
  _Float16* tgt16  = (_Float16*)alloc(NTOK * DM * 2);
  _Float16* mem16  = (_Float16*)alloc(NTOK * DM * 2);
  _Float16* qkv16  = (_Float16*)alloc(NTOK * DM * 2);
  _Float16* kv16   = (_Float16*)alloc(NTOK * DM * 2);
  _Float16* attn16 = (_Float16*)alloc(NTOK * DM * 2);
  float*    x32    = (float*)alloc(NTOK * DM * 4);
  _Float16* x16    = (_Float16*)alloc(NTOK * DM * 2);
  float*    t32    = (float*)alloc(NTOK * DM * 4);
  _Float16* h16    = (_Float16*)alloc(NTOK * FFND * 2);
  float*    relA   = (float*)alloc((size_t)BN * HN * LQN * 4);
  float*    relS   = (float*)alloc((size_t)BN * HN * LQN * 4);

  dim3 blk(256);
  auto cvt = [&](const float* s, _Float16* d, size_t n) {
    cvt_f32_f16_kernel<<<dim3(1024), blk, 0, stream>>>(s, d, (int)n);
  };
  cvt(saWq, saWq16, (size_t)DM * DM);
  cvt(saWo, saWo16, (size_t)DM * DM);
  cvt(caWq, caWq16, (size_t)DM * DM);
  cvt(caWo, caWo16, (size_t)DM * DM);
  cvt(W1, W1_16, (size_t)DM * FFND);
  cvt(W2, W2_16, (size_t)FFND * DM);
  cvt(tgt, tgt16, NTOK * DM);
  cvt(memx, mem16, NTOK * DM);

  dim3 g_proj(DM / 128, NTOK / 128);   // (8, 32)
  dim3 g_ffn1(FFND / 128, NTOK / 128); // (32, 32)
  dim3 g_attn(LQN / 128, HN, BN);      // (8, 16, 4)
  dim3 g_rel((BN * HN * LQN) / 256);
  dim3 g_ln(NTOK);

  // ---- self attention (q = k = v = tgt @ Wq + bq)
  gemm_wmma_kernel<false, false, true, false><<<g_proj, blk, 0, stream>>>(
      tgt16, saWq16, sabq, nullptr, nullptr, qkv16, (int)NTOK, DM, DM);
  relstats_kernel<<<g_rel, blk, 0, stream>>>(qkv16, relA, relS);
  attn_wmma_kernel<true><<<g_attn, blk, 0, stream>>>(qkv16, qkv16, relA, relS,
                                                     attn16);
  gemm_wmma_kernel<false, true, false, true><<<g_proj, blk, 0, stream>>>(
      attn16, saWo16, sabo, tgt, t32, nullptr, (int)NTOK, DM, DM);
  layernorm_kernel<<<g_ln, blk, 0, stream>>>(t32, ln1g, ln1b, x32, x16);

  // ---- cross attention (q = x @ Wq; k = v = memory @ Wq)
  gemm_wmma_kernel<false, false, true, false><<<g_proj, blk, 0, stream>>>(
      x16, caWq16, cabq, nullptr, nullptr, qkv16, (int)NTOK, DM, DM);
  gemm_wmma_kernel<false, false, true, false><<<g_proj, blk, 0, stream>>>(
      mem16, caWq16, cabq, nullptr, nullptr, kv16, (int)NTOK, DM, DM);
  relstats_kernel<<<g_rel, blk, 0, stream>>>(qkv16, relA, relS);
  attn_wmma_kernel<false><<<g_attn, blk, 0, stream>>>(qkv16, kv16, relA, relS,
                                                      attn16);
  gemm_wmma_kernel<false, true, false, true><<<g_proj, blk, 0, stream>>>(
      attn16, caWo16, cabo, x32, t32, nullptr, (int)NTOK, DM, DM);
  layernorm_kernel<<<g_ln, blk, 0, stream>>>(t32, ln2g, ln2b, x32, x16);

  // ---- FFN
  gemm_wmma_kernel<true, false, true, false><<<g_ffn1, blk, 0, stream>>>(
      x16, W1_16, b1, nullptr, nullptr, h16, (int)NTOK, FFND, DM);
  gemm_wmma_kernel<false, true, false, true><<<g_proj, blk, 0, stream>>>(
      h16, W2_16, b2, x32, t32, nullptr, (int)NTOK, DM, FFND);
  layernorm_kernel<<<g_ln, blk, 0, stream>>>(t32, ln3g, ln3b, out, nullptr);
}
